// TransformerEncoder_23278722744714
// MI455X (gfx1250) — matrix operand order repaired and checked by reading, not compiled, not run on hardware
//
#include <hip/hip_runtime.h>
#include <hip/hip_bf16.h>

// ---------------------------------------------------------------------------
// Types for CDNA5 WMMA (wave32): 16x16x32 bf16 -> f32 accum
// ---------------------------------------------------------------------------
typedef __attribute__((ext_vector_type(16))) __bf16 v16bf;
typedef __attribute__((ext_vector_type(8)))  float  v8f;

// ---------------------------------------------------------------------------
// CDNA5 async LDS copy: 16B global -> LDS, tracked on ASYNCcnt.
// lds = workgroup-relative LDS byte offset (low 32 bits of a flat &shared).
// ---------------------------------------------------------------------------
__device__ __forceinline__ void async_cp16(unsigned lds, const void* g) {
    asm volatile("global_load_async_to_lds_b128 %0, %1, off"
                 :: "v"(lds), "v"(g) : "memory");
}
__device__ __forceinline__ void wait_async0() {
    asm volatile("s_wait_asynccnt 0x0" ::: "memory");
}
__device__ __forceinline__ unsigned lds_off(const void* p) {
    return (unsigned)(unsigned long long)p;
}

// ---------------------------------------------------------------------------
// bf16 WMMA GEMM:  C[M,N] = alpha * (A[M,K] @ B^T + bias)
// A: bf16 [M,K] row-major (lda). B: bf16 [N,K] row-major (ldb) — always the
// "transposed" operand, so both tiles are K-contiguous pure copies staged with
// global_load_async_to_lds_b128. Batched over blockIdx.z.
// OUTBF: write bf16; OUTT: write transposed bf16 (V projection): row m of the
// output maps to batch b=m/ldc, s=m%ldc, element [b*sCb + n*ldc + s].
// Tile: BM=128 x BNt x BK=32, 256 threads = 8 waves, double-buffered LDS.
// ---------------------------------------------------------------------------
#define BM 128
#define BK 32
#define LPAD 8   // ushorts; LDS row stride 40 ushorts = 80 B (16B multiple)

template <int BNt, bool RELU, bool OUTBF, bool OUTT>
__global__ __launch_bounds__(256)
void gemm_bf16_wmma(const __bf16* __restrict__ A, long sAb, int lda,
                    const __bf16* __restrict__ Bt, long sBb, int ldb,
                    const float* __restrict__ bias,
                    void* __restrict__ Cv, long sCb, int ldc,
                    int K, float alpha)
{
    __shared__ __align__(16) unsigned short As[2][BM][BK + LPAD];
    __shared__ __align__(16) unsigned short Bs[2][BNt][BK + LPAD];

    const int tid   = threadIdx.x;
    const int lane  = tid & 31;
    const int wave  = tid >> 5;        // 0..7
    const int hlane = lane & 15;
    const int hhalf = lane >> 4;       // 0..1

    const int batch = blockIdx.z;
    A  += (long)batch * sAb;
    Bt += (long)batch * sBb;

    const int bm = blockIdx.y * BM;
    const int bn = blockIdx.x * BNt;

    // ---- async staging coordinates: each lane copies 16B chunks (8 bf16)
    // A tile: 128 rows x 64B -> 512 chunks -> 2 per thread
    const int a_row = tid >> 1;
    const int a_el  = (tid & 1) * 16;        // element offset 0 or 16
    const __bf16* a_gbase = A + (long)(bm + a_row) * lda + a_el;
    const unsigned a_l0 = lds_off(&As[0][a_row][a_el]);
    const unsigned a_l1 = lds_off(&As[1][a_row][a_el]);

    auto stageA = [&](int k0, int buf) {
        const __bf16* g = a_gbase + k0;
        const unsigned l = buf ? a_l1 : a_l0;
        async_cp16(l,      g);
        async_cp16(l + 16, g + 8);
    };

    // B tile: BNt rows x 64B
    auto stageB = [&](int k0, int buf) {
        if (BNt == 128) {                    // 2 chunks per thread
            const int n  = tid >> 1;
            const int el = (tid & 1) * 16;
            const __bf16* g = Bt + (long)(bn + n) * ldb + k0 + el;
            const unsigned l = lds_off(&Bs[buf][n][el]);
            async_cp16(l,      g);
            async_cp16(l + 16, g + 8);
        } else {                             // BNt==64: 1 chunk per thread
            const int n  = tid >> 2;
            const int el = (tid & 3) * 8;
            const __bf16* g = Bt + (long)(bn + n) * ldb + k0 + el;
            async_cp16(lds_off(&Bs[buf][n][el]), g);
        }
    };

    // ---- accumulators (bias folded in)
    v8f acc[BNt / 16];
#pragma unroll
    for (int t = 0; t < BNt / 16; ++t) {
        float bv = bias ? bias[bn + t * 16 + hlane] : 0.0f;
#pragma unroll
        for (int i = 0; i < 8; ++i) acc[t][i] = bv;
    }

    const int m_frag = wave * 16 + hlane;
    const int a_koff = hhalf * 16;
    const int b_koff = hhalf * 16;

    auto compute = [&](int cur) {
        union { v16bf v; uint4 q[2]; } afrag;
        afrag.q[0] = *reinterpret_cast<const uint4*>(&As[cur][m_frag][a_koff]);
        afrag.q[1] = *reinterpret_cast<const uint4*>(&As[cur][m_frag][a_koff + 8]);
        union { v16bf v; uint4 q[2]; } bfrag[BNt / 16];
#pragma unroll
        for (int t = 0; t < BNt / 16; ++t) {
            const unsigned short* bp = &Bs[cur][t * 16 + hlane][b_koff];
            bfrag[t].q[0] = reinterpret_cast<const uint4*>(bp)[0];
            bfrag[t].q[1] = reinterpret_cast<const uint4*>(bp)[1];
        }
#pragma unroll
        for (int t = 0; t < BNt / 16; ++t)
            acc[t] = __builtin_amdgcn_wmma_f32_16x16x32_bf16(
                false, afrag.v, false, bfrag[t].v, (short)0, acc[t], false, false);
    };

    // ---- prologue: async-stage k-step 0
    stageA(0, 0);
    stageB(0, 0);
    wait_async0();
    __syncthreads();

    const int ksteps = K / BK;               // >= 2 for all call sites
    for (int ks = 0; ks < ksteps - 1; ++ks) {
        const int cur = ks & 1;
        stageA((ks + 1) * BK, cur ^ 1);      // async engine fills next buffer
        stageB((ks + 1) * BK, cur ^ 1);
        compute(cur);                         // WMMA burst overlaps the DMA
        wait_async0();
        __syncthreads();
    }
    compute((ksteps - 1) & 1);

    // ---- epilogue (C/D layout: row = vgpr + 8*hhalf, col = hlane)
#pragma unroll
    for (int t = 0; t < BNt / 16; ++t) {
#pragma unroll
        for (int vv = 0; vv < 8; ++vv) {
            const int m = bm + wave * 16 + vv + 8 * hhalf;
            const int n = bn + t * 16 + hlane;
            float r = acc[t][vv] * alpha;
            if (RELU) r = fmaxf(r, 0.0f);
            if (OUTT) {
                const int bb = m / ldc;       // ldc = S here
                const int sr = m - bb * ldc;
                ((__bf16*)Cv)[(long)bb * sCb + (long)n * ldc + sr] = (__bf16)r;
            } else if (OUTBF) {
                ((__bf16*)Cv)[(long)batch * sCb + (long)m * ldc + n] = (__bf16)r;
            } else {
                ((float*)Cv)[(long)batch * sCb + (long)m * ldc + n] = r;
            }
        }
    }
}

// ---------------------------------------------------------------------------
// Transpose + fp32->bf16 convert: in [K,N] f32 (z-batched) -> out [N,K] bf16
// ---------------------------------------------------------------------------
__global__ __launch_bounds__(256)
void transpose_cvt_kernel(const float* __restrict__ in, __bf16* __restrict__ out,
                          int K, int N)
{
    __shared__ float tile[32][33];
    const long lz = (long)blockIdx.z * K * N;
    const int k0 = blockIdx.y * 32, n0 = blockIdx.x * 32;
    const int tx = threadIdx.x & 31, ty = threadIdx.x >> 5;   // 32 x 8
#pragma unroll
    for (int r = 0; r < 32; r += 8)
        tile[ty + r][tx] = in[lz + (long)(k0 + ty + r) * N + n0 + tx];
    __syncthreads();
#pragma unroll
    for (int r = 0; r < 32; r += 8)
        out[lz + (long)(n0 + ty + r) * K + k0 + tx] = (__bf16)tile[tx][ty + r];
}

// ---------------------------------------------------------------------------
// Block reductions (wave32)
// ---------------------------------------------------------------------------
__device__ __forceinline__ float block_sum(float v, float* red) {
#pragma unroll
    for (int o = 16; o > 0; o >>= 1) v += __shfl_down(v, o, 32);
    const int lane = threadIdx.x & 31, w = threadIdx.x >> 5;
    if (lane == 0) red[w] = v;
    __syncthreads();
    float r = (threadIdx.x < 8) ? red[threadIdx.x] : 0.0f;
    if (w == 0) {
#pragma unroll
        for (int o = 4; o > 0; o >>= 1) r += __shfl_down(r, o, 32);
        if (lane == 0) red[0] = r;
    }
    __syncthreads();
    r = red[0];
    __syncthreads();
    return r;
}

__device__ __forceinline__ float block_max(float v, float* red) {
#pragma unroll
    for (int o = 16; o > 0; o >>= 1) v = fmaxf(v, __shfl_down(v, o, 32));
    const int lane = threadIdx.x & 31, w = threadIdx.x >> 5;
    if (lane == 0) red[w] = v;
    __syncthreads();
    float r = (threadIdx.x < 8) ? red[threadIdx.x] : -3.0e38f;
    if (w == 0) {
#pragma unroll
        for (int o = 4; o > 0; o >>= 1) r = fmaxf(r, __shfl_down(r, o, 32));
        if (lane == 0) red[0] = r;
    }
    __syncthreads();
    r = red[0];
    __syncthreads();
    return r;
}

// ---------------------------------------------------------------------------
// LayerNorm: fp32 out (residual chain) + optional bf16 out (GEMM A operand)
// ---------------------------------------------------------------------------
__global__ __launch_bounds__(256)
void ln_kernel(const float* __restrict__ x, const float* __restrict__ g,
               const float* __restrict__ b, float* __restrict__ y,
               __bf16* __restrict__ yb, int D_)
{
    __shared__ float red[8];
    const long row = blockIdx.x;
    const float* xr = x + row * D_;

    float s = 0.0f, s2 = 0.0f;
    for (int i = threadIdx.x; i < D_; i += blockDim.x) {
        float v = xr[i]; s += v; s2 += v * v;
    }
    s  = block_sum(s, red);
    s2 = block_sum(s2, red);
    const float mu  = s / (float)D_;
    const float var = s2 / (float)D_ - mu * mu;
    const float inv = rsqrtf(var + 1e-6f);
    for (int i = threadIdx.x; i < D_; i += blockDim.x) {
        float r = (xr[i] - mu) * inv * g[i] + b[i];
        y[row * D_ + i] = r;
        if (yb) yb[row * D_ + i] = (__bf16)r;
    }
}

// ---------------------------------------------------------------------------
// Masked softmax: fp32 scores [B,S,S] -> bf16 probs. grid = (S, B)
// ---------------------------------------------------------------------------
__global__ __launch_bounds__(256)
void softmax_mask_kernel(const float* __restrict__ w, __bf16* __restrict__ p,
                         const int* __restrict__ npm, int S_)
{
    __shared__ float red[8];
    const long rb = ((long)blockIdx.y * S_ + blockIdx.x) * S_;
    const float* row = w + rb;
    __bf16* prow = p + rb;
    const int* mrow = npm + (long)blockIdx.y * S_;

    float mx = -3.0e38f;
    for (int i = threadIdx.x; i < S_; i += blockDim.x) {
        float v = (mrow[i] == 0) ? -1.0e9f : row[i];
        mx = fmaxf(mx, v);
    }
    mx = block_max(mx, red);

    float sum = 0.0f;
    for (int i = threadIdx.x; i < S_; i += blockDim.x) {
        float v = (mrow[i] == 0) ? -1.0e9f : row[i];
        sum += __expf(v - mx);
    }
    sum = block_sum(sum, red);
    const float inv = 1.0f / sum;
    for (int i = threadIdx.x; i < S_; i += blockDim.x) {
        float v = (mrow[i] == 0) ? -1.0e9f : row[i];
        prow[i] = (__bf16)(__expf(v - mx) * inv);
    }
}

// ---------------------------------------------------------------------------
// Elementwise kernels
// ---------------------------------------------------------------------------
__global__ __launch_bounds__(256)
void add_pe_kernel(const float* __restrict__ e, const float* __restrict__ pe,
                   float* __restrict__ x, int S_, int D_)
{
    const long i = (long)blockIdx.x * blockDim.x + threadIdx.x;
    const int  d = (int)(i % D_);
    const long s = (i / D_) % S_;
    x[i] = e[i] + pe[s * D_ + d];
}

__global__ __launch_bounds__(256)
void resid_mask_kernel(const float* __restrict__ a, const float* __restrict__ b,
                       const int* __restrict__ npm, float* __restrict__ y, int D_)
{
    const long i = (long)blockIdx.x * blockDim.x + threadIdx.x;
    const long row = i / D_;
    const float m = (float)npm[row];
    y[i] = (a[i] + b[i]) * m;
}

// ---------------------------------------------------------------------------
// Host launcher
// ---------------------------------------------------------------------------
extern "C" void kernel_launch(void* const* d_in, const int* in_sizes, int n_in,
                              void* d_out, int out_size, void* d_ws, size_t ws_size,
                              hipStream_t stream)
{
    (void)in_sizes; (void)n_in; (void)out_size; (void)ws_size;

    constexpr int Bz = 4, S = 1024, D = 1024, Hh = 16, DK = 64, F = 4096, L = 6;
    const float SCALE = 0.125f;   // 1/sqrt(64)

    const float* embed = (const float*)d_in[0];
    const int*   npm   = (const int*)  d_in[1];
    const float* pe    = (const float*)d_in[2];
    const float* Wq = (const float*)d_in[3];  const float* bq = (const float*)d_in[4];
    const float* Wk = (const float*)d_in[5];  const float* bk = (const float*)d_in[6];
    const float* Wv = (const float*)d_in[7];  const float* bv = (const float*)d_in[8];
    const float* Wo = (const float*)d_in[9];  const float* bo = (const float*)d_in[10];
    const float* g1 = (const float*)d_in[11]; const float* be1 = (const float*)d_in[12];
    const float* W1 = (const float*)d_in[13]; const float* c1 = (const float*)d_in[14];
    const float* W2 = (const float*)d_in[15]; const float* c2 = (const float*)d_in[16];
    const float* g2 = (const float*)d_in[17]; const float* be2 = (const float*)d_in[18];
    const float* gf = (const float*)d_in[19]; const float* bf = (const float*)d_in[20];
    float* out = (float*)d_out;

    const long BSD = (long)Bz * S * D;     // 4 Mi elems
    const long BSS = (long)Bz * S * S;
    const long BSF = (long)Bz * S * F;

    // scratch arena
    char* base = (char*)d_ws; size_t off = 0;
    auto allocf = [&](long nelem) -> float* {
        float* p = (float*)(base + off);
        off = (off + (size_t)nelem * 4 + 255) & ~(size_t)255;
        return p;
    };
    auto allocb = [&](long nelem) -> __bf16* {
        __bf16* p = (__bf16*)(base + off);
        off = (off + (size_t)nelem * 2 + 255) & ~(size_t)255;
        return p;
    };
    // fp32 residual chain
    float* x      = allocf(BSD);   // x / h2 (per-layer rotation)
    float* hbuf   = allocf(BSD);   // h / x1
    float* scores = allocf(BSS);   // scores / att / ff (aliased lifetimes)
    // bf16 activations
    __bf16* hb16  = allocb(BSD);   // h / h2 bf16
    __bf16* qb16  = allocb(BSD);
    __bf16* kb16  = allocb(BSD);
    __bf16* vt16  = allocb(BSD);   // V transposed: [B,H,DK,S]
    __bf16* ob16  = allocb(BSD);
    __bf16* pb16  = allocb(BSS);   // softmax probs (per head loop)
    __bf16* f116  = allocb(BSF);
    // bf16 transposed weights [N,K] per layer
    __bf16* wqT = allocb((long)L * D * D);
    __bf16* wkT = allocb((long)L * D * D);
    __bf16* wvT = allocb((long)L * D * D);
    __bf16* woT = allocb((long)L * D * D);
    __bf16* w1T = allocb((long)L * D * F);
    __bf16* w2T = allocb((long)L * F * D);

    // ---- one-time weight transpose+convert (L batched in z)
    transpose_cvt_kernel<<<dim3(D / 32, D / 32, L), 256, 0, stream>>>(Wq, wqT, D, D);
    transpose_cvt_kernel<<<dim3(D / 32, D / 32, L), 256, 0, stream>>>(Wk, wkT, D, D);
    transpose_cvt_kernel<<<dim3(D / 32, D / 32, L), 256, 0, stream>>>(Wv, wvT, D, D);
    transpose_cvt_kernel<<<dim3(D / 32, D / 32, L), 256, 0, stream>>>(Wo, woT, D, D);
    transpose_cvt_kernel<<<dim3(F / 32, D / 32, L), 256, 0, stream>>>(W1, w1T, D, F);
    transpose_cvt_kernel<<<dim3(D / 32, F / 32, L), 256, 0, stream>>>(W2, w2T, F, D);

    const int EW_BLOCKS = (int)(BSD / 256);
    add_pe_kernel<<<EW_BLOCKS, 256, 0, stream>>>(embed, pe, x, S, D);

    const dim3 gProj(D / 128, (Bz * S) / BM, 1);      // (8, 32)
    const dim3 gF1  (F / 128, (Bz * S) / BM, 1);      // (32, 32)
    const dim3 gSc  (S / 128, S / BM, Bz);            // (8, 8, 4)
    const dim3 gAV  (DK / 64, S / BM, Bz);            // (1, 8, 4)

    for (int l = 0; l < L; ++l) {
        const __bf16* wq = wqT + (long)l * D * D;  const float* bq_ = bq + (long)l * D;
        const __bf16* wk = wkT + (long)l * D * D;  const float* bk_ = bk + (long)l * D;
        const __bf16* wv = wvT + (long)l * D * D;  const float* bv_ = bv + (long)l * D;
        const __bf16* wo = woT + (long)l * D * D;  const float* bo_ = bo + (long)l * D;
        const __bf16* w1 = w1T + (long)l * D * F;  const float* c1_ = c1 + (long)l * F;
        const __bf16* w2 = w2T + (long)l * F * D;  const float* c2_ = c2 + (long)l * D;

        // h = LN(x) -> hbuf (fp32) + hb16 (bf16)
        ln_kernel<<<Bz * S, 256, 0, stream>>>(x, g1 + (long)l * D, be1 + (long)l * D,
                                              hbuf, hb16, D);

        // q,k bf16; v transposed bf16 [B,H,DK,S]
        gemm_bf16_wmma<128, false, true, false><<<gProj, 256, 0, stream>>>(
            hb16, 0, D, wq, 0, D, bq_, qb16, 0, D, D, 1.0f);
        gemm_bf16_wmma<128, false, true, false><<<gProj, 256, 0, stream>>>(
            hb16, 0, D, wk, 0, D, bk_, kb16, 0, D, D, 1.0f);
        gemm_bf16_wmma<128, false, true, true><<<gProj, 256, 0, stream>>>(
            hb16, 0, D, wv, 0, D, bv_, vt16, (long)D * S, S, D, 1.0f);

        // attention, one head at a time, batched over B in z
        for (int hh = 0; hh < Hh; ++hh) {
            // scores = SCALE * Q @ K^T (fp32)
            gemm_bf16_wmma<128, false, false, false><<<gSc, 256, 0, stream>>>(
                qb16 + hh * DK, (long)S * D, D,
                kb16 + hh * DK, (long)S * D, D,
                nullptr,
                scores, (long)S * S, S,
                DK, SCALE);
            // masked softmax -> bf16 probs
            softmax_mask_kernel<<<dim3(S, Bz), 256, 0, stream>>>(scores, pb16, npm, S);
            // O = P @ V  (B operand = Vt rows, K-contiguous)
            gemm_bf16_wmma<64, false, true, false><<<gAV, 256, 0, stream>>>(
                pb16, (long)S * S, S,
                vt16 + (long)hh * DK * S, (long)D * S, S,
                nullptr,
                ob16 + hh * DK, (long)S * D, D,
                S, 1.0f);
        }

        // att = O @ Wo + bo (fp32) -> scores buffer
        gemm_bf16_wmma<128, false, false, false><<<gProj, 256, 0, stream>>>(
            ob16, 0, D, wo, 0, D, bo_, scores, 0, D, D, 1.0f);

        // x1 = (h + att) * mask -> hbuf (in-place safe)
        resid_mask_kernel<<<EW_BLOCKS, 256, 0, stream>>>(hbuf, scores, npm, hbuf, D);

        // h2 = LN(x1) -> x (fp32) + hb16 (bf16)
        ln_kernel<<<Bz * S, 256, 0, stream>>>(hbuf, g2 + (long)l * D, be2 + (long)l * D,
                                              x, hb16, D);

        // ff1 = relu(h2 @ W1 + c1) -> bf16
        gemm_bf16_wmma<128, true, true, false><<<gF1, 256, 0, stream>>>(
            hb16, 0, D, w1, 0, D, c1_, f116, 0, F, D, 1.0f);

        // ff = ff1 @ W2 + c2 (fp32) -> scores buffer
        gemm_bf16_wmma<128, false, false, false><<<gProj, 256, 0, stream>>>(
            f116, 0, F, w2, 0, F, c2_, scores, 0, D, F, 1.0f);

        // x = (h2 + ff) * mask (in-place safe)
        resid_mask_kernel<<<EW_BLOCKS, 256, 0, stream>>>(x, scores, npm, x, D);
    }

    // out = LN(x, lnf_g, lnf_b)
    ln_kernel<<<Bz * S, 256, 0, stream>>>(x, gf, bf, out, nullptr, D);
}
